// RelativeSelfAttention_30889404793233
// MI455X (gfx1250) — matrix operand order repaired and checked by reading, not compiled, not run on hardware
//
#include <hip/hip_runtime.h>
#include <stdint.h>

typedef __attribute__((ext_vector_type(16))) _Float16 v16h;
typedef __attribute__((ext_vector_type(8)))  _Float16 v8h;
typedef __attribute__((ext_vector_type(8)))  float    v8f;

#define T_SEQ   2048
#define BATCH   2
#define DMODEL  1024
#define NHEAD   16
#define DHEAD   64
#define BPAD    80   // padded B-stage row (halves); 160B row stride, 32B aligned

static __device__ __forceinline__ v16h cat8(v8h lo, v8h hi) {
  return __builtin_shufflevector(lo, hi, 0,1,2,3,4,5,6,7,8,9,10,11,12,13,14,15);
}
static __device__ __forceinline__ v8f wmma_f16f32(v16h a, v16h b, v8f c) {
  // D = A(16xK32) * B(K32x16) + C, fp32 accumulate
  return __builtin_amdgcn_wmma_f32_16x16x32_f16(false, a, false, b, (short)0, c, false, false);
}
static __device__ __forceinline__ v8f zero8() { v8f r = {}; return r; }

// ---- CDNA5 async copy: global -> LDS, tracked by ASYNCcnt (ISA 15.18.3) ----
static __device__ __forceinline__ void async_b128(const void* gaddr, uint32_t ldsoff) {
  asm volatile("global_load_async_to_lds_b128 %0, %1, off"
               :: "v"(ldsoff), "v"((unsigned long long)(uintptr_t)gaddr)
               : "memory");
}
static __device__ __forceinline__ void wait_async0() {
  asm volatile("s_wait_asynccnt 0x0" ::: "memory");
}
// low 32 bits of a generic pointer to __shared__ == wave-relative LDS address
static __device__ __forceinline__ uint32_t lds_off(const void* p) {
  return (uint32_t)(uintptr_t)p;
}

// ---------------------------------------------------------------- utilities
__global__ void cvt_kernel(const float* __restrict__ src, _Float16* __restrict__ dst, int n) {
  int i = blockIdx.x * 256 + threadIdx.x;
  if (i < n) dst[i] = (_Float16)src[i];
}

// LayerNorm over D=1024, one row per block (row = t*B+b), output f16
__global__ void __launch_bounds__(256)
ln_kernel(const float* __restrict__ x, const float* __restrict__ g,
          const float* __restrict__ bt, _Float16* __restrict__ out) {
  __shared__ float sred[2][8];
  const int row = blockIdx.x;
  const float* xr = x + (size_t)row * DMODEL;
  float v[4], s = 0.f, ss = 0.f;
#pragma unroll
  for (int k = 0; k < 4; ++k) {
    v[k] = xr[threadIdx.x + 256 * k];
    s += v[k]; ss += v[k] * v[k];
  }
#pragma unroll
  for (int off = 16; off >= 1; off >>= 1) {
    s  += __shfl_xor(s,  off, 32);
    ss += __shfl_xor(ss, off, 32);
  }
  const int wid = threadIdx.x >> 5;
  if ((threadIdx.x & 31) == 0) { sred[0][wid] = s; sred[1][wid] = ss; }
  __syncthreads();
  s = 0.f; ss = 0.f;
#pragma unroll
  for (int w = 0; w < 8; ++w) { s += sred[0][w]; ss += sred[1][w]; }
  const float mu  = s * (1.f / DMODEL);
  const float var = ss * (1.f / DMODEL) - mu * mu;
  const float rs  = rsqrtf(var + 1e-5f);
#pragma unroll
  for (int k = 0; k < 4; ++k) {
    const int d = threadIdx.x + 256 * k;
    out[(size_t)row * DMODEL + d] = (_Float16)((v[k] - mu) * rs * g[d] + bt[d]);
  }
}

// ---------------------------------------------------------------- GEMM core
// 4-wave block computes 64(M) x 64(N). The 32x64 B chunk is staged in LDS via
// double-buffered GLOBAL_LOAD_ASYNC_TO_LDS_B128 (ASYNCcnt), shared by all 4
// waves (4x reduction of B global traffic). Each wave owns 16 rows of A/C.
static __device__ __forceinline__ void gemm_64x64_async(
    const _Float16* __restrict__ A, int lda,   // pre-offset to block row m0
    const _Float16* __restrict__ B, int ldb,   // pre-offset to block col n0
    int K, _Float16 (*Blds)[32][BPAD], v8f acc[4]) {
  const int tid  = threadIdx.x;
  const int lane = tid & 31, u = lane & 15, g = lane >> 4, wid = tid >> 5;
  const _Float16* arow = A + (size_t)(16 * wid + u) * lda;

  // this thread's 16-half slice of the 32x64 stage
  const int srow = tid >> 2;
  const int scol = (tid & 3) * 16;
  const int niter = K >> 5;

  {  // prologue: stage kc=0 into buffer 0
    const _Float16* gp = B + (size_t)srow * ldb + scol;
    const uint32_t  lo = lds_off(&Blds[0][srow][scol]);
    async_b128(gp, lo);
    async_b128(gp + 8, lo + 16);
  }
  for (int i = 0; i < niter; ++i) {
    const int kc = i << 5;
    const int cur = i & 1;
    wait_async0();        // own slice of buffer `cur` has landed
    __syncthreads();      // everyone's slice landed; buffer cur^1 free to fill
    if (i + 1 < niter) {
      const _Float16* gp = B + (size_t)(kc + 32 + srow) * ldb + scol;
      const uint32_t  lo = lds_off(&Blds[cur ^ 1][srow][scol]);
      async_b128(gp, lo);       // overlaps with the WMMAs below
      async_b128(gp + 8, lo + 16);
    }
    v16h af = cat8(*(const v8h*)(arow + kc + 8 * g),
                   *(const v8h*)(arow + kc + 16 + 8 * g));
#pragma unroll
    for (int nt = 0; nt < 4; ++nt) {
      v16h bf;
#pragma unroll
      for (int s = 0; s < 8; ++s) {
        bf[s]     = Blds[cur][8 * g + s][16 * nt + u];
        bf[s + 8] = Blds[cur][16 + 8 * g + s][16 * nt + u];
      }
      acc[nt] = wmma_f16f32(af, bf, acc[nt]);
    }
  }
}

// qkv = xln[4096,1024] x Wqkv[1024,3072]; scatter to q/k/v [b][h][t][64] f16
__global__ void __launch_bounds__(128)
gemm_qkv_kernel(const _Float16* __restrict__ A, const _Float16* __restrict__ W,
                _Float16* __restrict__ qkv) {
  __shared__ alignas(32) _Float16 Blds[2][32][BPAD];
  const int m0 = blockIdx.x * 64, n0 = blockIdx.y * 64;
  v8f acc[4] = { zero8(), zero8(), zero8(), zero8() };
  gemm_64x64_async(A + (size_t)m0 * DMODEL, DMODEL, W + n0, 3 * DMODEL, DMODEL, Blds, acc);
  const int lane = threadIdx.x & 31, u = lane & 15, g = lane >> 4, wid = threadIdx.x >> 5;
#pragma unroll
  for (int nt = 0; nt < 4; ++nt)
#pragma unroll
    for (int r = 0; r < 8; ++r) {
      const int row = m0 + 16 * wid + 8 * g + r;
      const int t = row >> 1, b = row & 1;
      const int e = n0 + 16 * nt + u;
      const int s = e >> 10, hh = (e >> 6) & 15, dh = e & 63;
      qkv[(((size_t)(s * BATCH + b) * NHEAD + hh) * T_SEQ + t) * DHEAD + dh] =
          (_Float16)acc[nt][r];
    }
}

// r = pos[2048,1024] x Wr[1024,1024]; scatter to r_f16[h][t][64]
__global__ void __launch_bounds__(128)
gemm_r_kernel(const _Float16* __restrict__ A, const _Float16* __restrict__ W,
              _Float16* __restrict__ rf) {
  __shared__ alignas(32) _Float16 Blds[2][32][BPAD];
  const int m0 = blockIdx.x * 64, n0 = blockIdx.y * 64;
  v8f acc[4] = { zero8(), zero8(), zero8(), zero8() };
  gemm_64x64_async(A + (size_t)m0 * DMODEL, DMODEL, W + n0, DMODEL, DMODEL, Blds, acc);
  const int lane = threadIdx.x & 31, u = lane & 15, g = lane >> 4, wid = threadIdx.x >> 5;
#pragma unroll
  for (int nt = 0; nt < 4; ++nt)
#pragma unroll
    for (int r = 0; r < 8; ++r) {
      const int t = m0 + 16 * wid + 8 * g + r;
      const int e = n0 + 16 * nt + u;
      const int hh = e >> 6, dh = e & 63;
      rf[((size_t)hh * T_SEQ + t) * DHEAD + dh] = (_Float16)acc[nt][r];
    }
}

// out = attn[4096,1024] x Wo[1024,1024] -> fp32 [T,B,D]
__global__ void __launch_bounds__(128)
gemm_out_kernel(const _Float16* __restrict__ A, const _Float16* __restrict__ W,
                float* __restrict__ out) {
  __shared__ alignas(32) _Float16 Blds[2][32][BPAD];
  const int m0 = blockIdx.x * 64, n0 = blockIdx.y * 64;
  v8f acc[4] = { zero8(), zero8(), zero8(), zero8() };
  gemm_64x64_async(A + (size_t)m0 * DMODEL, DMODEL, W + n0, DMODEL, DMODEL, Blds, acc);
  const int lane = threadIdx.x & 31, u = lane & 15, g = lane >> 4, wid = threadIdx.x >> 5;
#pragma unroll
  for (int nt = 0; nt < 4; ++nt)
#pragma unroll
    for (int r = 0; r < 8; ++r) {
      const int row = m0 + 16 * wid + 8 * g + r;
      out[(size_t)row * DMODEL + n0 + 16 * nt + u] = acc[nt][r];
    }
}

// -------------------------------------------------- fused causal attention
// One wave per 16-query tile of one (b,h). Scores computed transposed so the
// softmax P C-fragments feed the PV WMMA A-operand with zero relayout.
// rel_shift handled as a 48-wide banded WMMA against r + LDS diagonal gather.
__global__ void __launch_bounds__(32)
attn_kernel(const _Float16* __restrict__ qf, const _Float16* __restrict__ kf,
            const _Float16* __restrict__ vf, const _Float16* __restrict__ rf,
            const float* __restrict__ rwb, const float* __restrict__ rrb,
            _Float16* __restrict__ attn) {
  const int i0 = blockIdx.x * 16;
  const int h  = blockIdx.y;
  const int b  = blockIdx.z;
  const int lane = threadIdx.x & 31;
  const int u = lane & 15;
  const int g = lane >> 4;

  __shared__ alignas(32) _Float16 Qwbt[64][16];   // (q + r_w_bias)^T
  __shared__ alignas(32) _Float16 Qrbt[64][16];   // (q + r_r_bias)^T
  __shared__ float Wlds[48][16];                  // BD band scratch

  const size_t headT = (size_t)(b * NHEAD + h) * T_SEQ;
  const _Float16* qb = qf + headT * DHEAD;
  const _Float16* kb = kf + headT * DHEAD;
  const _Float16* vb = vf + headT * DHEAD;
  const _Float16* rb = rf + (size_t)h * T_SEQ * DHEAD;

  // stage Q^T with both biases (once per query tile)
  for (int idx = lane; idx < 16 * DHEAD; idx += 32) {
    const int uu = idx >> 6, d = idx & 63;
    const float qv = (float)qb[(size_t)(i0 + uu) * DHEAD + d];
    Qwbt[d][uu] = (_Float16)(qv + rwb[h * DHEAD + d]);
    Qrbt[d][uu] = (_Float16)(qv + rrb[h * DHEAD + d]);
  }
  __syncthreads();

  v16h qwbf[2], qrbf[2];            // B-fragments: lane = query, elements = dh (table order)
#pragma unroll
  for (int c = 0; c < 2; ++c) {
#pragma unroll
    for (int s = 0; s < 8; ++s) {
      qwbf[c][s]     = Qwbt[32 * c + 8 * g + s][u];
      qwbf[c][s + 8] = Qwbt[32 * c + 16 + 8 * g + s][u];
      qrbf[c][s]     = Qrbt[32 * c + 8 * g + s][u];
      qrbf[c][s + 8] = Qrbt[32 * c + 16 + 8 * g + s][u];
    }
  }

  v8f o0 = zero8(), o1 = zero8(), o2 = zero8(), o3 = zero8();
  float mrun = -1e30f, lrun = 0.f;

  const int nchunk = ((i0 + 15) >> 5) + 1;
  for (int ch = 0; ch < nchunk; ++ch) {
    const int j0 = ch << 5;

    // AC term: S^T[key, query] = K * Q_wb^T
    v8f sacc[2] = { zero8(), zero8() };
#pragma unroll
    for (int s = 0; s < 2; ++s) {
      const _Float16* krow = kb + (size_t)(j0 + 16 * s + u) * DHEAD;
#pragma unroll
      for (int c = 0; c < 2; ++c) {
        v16h ka = cat8(*(const v8h*)(krow + 32 * c + 8 * g),
                       *(const v8h*)(krow + 32 * c + 16 + 8 * g));
        sacc[s] = wmma_f16f32(ka, qwbf[c], sacc[s]);
      }
    }

    // BD band: W^T[band_row, query] = R_band * Q_rb^T;  m = mstart + band_row
    const int mstart = T_SEQ - 16 - i0 + j0;   // c0 - 15, c0 = T-1-i0+j0
#pragma unroll
    for (int t = 0; t < 3; ++t) {
      int mrow = mstart + 16 * t + u;
      mrow = mrow < 0 ? 0 : (mrow > T_SEQ - 1 ? T_SEQ - 1 : mrow);
      const _Float16* rrow = rb + (size_t)mrow * DHEAD;
      v8f w = zero8();
#pragma unroll
      for (int c = 0; c < 2; ++c) {
        v16h ra = cat8(*(const v8h*)(rrow + 32 * c + 8 * g),
                       *(const v8h*)(rrow + 32 * c + 16 + 8 * g));
        w = wmma_f16f32(ra, qrbf[c], w);
      }
#pragma unroll
      for (int r = 0; r < 8; ++r) Wlds[16 * t + 8 * g + r][u] = w[r];
    }
    __syncthreads();

    if (ch + 1 < nchunk) {         // prefetch next K/V chunk -> global_prefetch_b8
      __builtin_prefetch(kb + (size_t)(j0 + 32 + lane) * DHEAD, 0, 1);
      __builtin_prefetch(vb + (size_t)(j0 + 32 + lane) * DHEAD, 0, 1);
    }

    // assemble scores: scale*(AC + rel-shifted BD), causal mask
#pragma unroll
    for (int s = 0; s < 2; ++s)
#pragma unroll
      for (int r = 0; r < 8; ++r) {
        const int v = 16 * s + 8 * g + r;          // key index in chunk
        const int j = j0 + v;
        const float sc = (sacc[s][r] + Wlds[15 + v - u][u]) * 0.125f;
        sacc[s][r] = (j > i0 + u) ? -1e30f : sc;
      }
    __syncthreads();               // Wlds reused next chunk

    // flash-softmax update (per-query stats in-lane; merge half-lanes)
    float mloc = mrun;
#pragma unroll
    for (int s = 0; s < 2; ++s)
#pragma unroll
      for (int r = 0; r < 8; ++r) mloc = fmaxf(mloc, sacc[s][r]);
    mloc = fmaxf(mloc, __shfl_xor(mloc, 16, 32));

    const float alpha = __expf(mrun - mloc);
    float pv[2][8], psum = 0.f;
#pragma unroll
    for (int s = 0; s < 2; ++s)
#pragma unroll
      for (int r = 0; r < 8; ++r) {
        const float p = __expf(sacc[s][r] - mloc);
        pv[s][r] = p; psum += p;
      }
    psum += __shfl_xor(psum, 16, 32);
    lrun = lrun * alpha + psum;
    mrun = mloc;

#pragma unroll
    for (int r = 0; r < 8; ++r) {  // rescale O rows (row = 8g+r lives in lane 8g+r)
      const float ar = __shfl(alpha, 8 * g + r, 32);
      o0[r] *= ar; o1[r] *= ar; o2[r] *= ar; o3[r] *= ar;
    }

    // P^T C-frags map directly onto the PV A-fragment layout
    v16h pa;
#pragma unroll
    for (int e = 0; e < 8; ++e) {
      pa[e]     = (_Float16)pv[0][e];
      pa[e + 8] = (_Float16)pv[1][e];
    }
    const _Float16* vcol = vb + (size_t)j0 * DHEAD + u;
    v16h vb0, vb1, vb2, vb3;
#pragma unroll
    for (int s = 0; s < 16; ++s) {
      const int kk = (s < 8) ? (8 * g + s) : (8 + 8 * g + s);
      const _Float16* vp = vcol + (size_t)kk * DHEAD;
      vb0[s] = vp[0]; vb1[s] = vp[16]; vb2[s] = vp[32]; vb3[s] = vp[48];
    }
    o0 = wmma_f16f32(pa, vb0, o0);
    o1 = wmma_f16f32(pa, vb1, o1);
    o2 = wmma_f16f32(pa, vb2, o2);
    o3 = wmma_f16f32(pa, vb3, o3);
  }

  // epilogue: divide by l, store attn_vec f16 rows (t*B+b, h*64+n)
#pragma unroll
  for (int r = 0; r < 8; ++r) {
    const float inv = 1.f / __shfl(lrun, 8 * g + r, 32);
    const int t = i0 + 8 * g + r;
    _Float16* orow = attn + ((size_t)t * BATCH + b) * DMODEL + h * DHEAD;
    orow[u]      = (_Float16)(o0[r] * inv);
    orow[u + 16] = (_Float16)(o1[r] * inv);
    orow[u + 32] = (_Float16)(o2[r] * inv);
    orow[u + 48] = (_Float16)(o3[r] * inv);
  }
}

// ---------------------------------------------------------------- launcher
extern "C" void kernel_launch(void* const* d_in, const int* in_sizes, int n_in,
                              void* d_out, int out_size, void* d_ws, size_t ws_size,
                              hipStream_t stream) {
  (void)in_sizes; (void)n_in; (void)out_size; (void)ws_size;
  const float* x    = (const float*)d_in[0];
  const float* pos  = (const float*)d_in[1];
  const float* ln_g = (const float*)d_in[2];
  const float* ln_b = (const float*)d_in[3];
  const float* Wqkv = (const float*)d_in[4];
  const float* Wr   = (const float*)d_in[5];
  const float* Wo   = (const float*)d_in[6];
  const float* rwb  = (const float*)d_in[7];
  const float* rrb  = (const float*)d_in[8];
  // d_in[9] = attn_mask (causal, baked into the kernel)

  char* ws = (char*)d_ws;                         // workspace layout (bytes)
  _Float16* xln   = (_Float16*)(ws + 0);          // 4096x1024 f16      (8 MB)
  _Float16* wqkvh = (_Float16*)(ws + 8388608);    // 1024x3072 f16      (6 MB)
  _Float16* wrh   = (_Float16*)(ws + 14680064);   // 1024x1024 f16      (2 MB)
  _Float16* woh   = (_Float16*)(ws + 16777216);   // 1024x1024 f16      (2 MB)
  _Float16* posh  = (_Float16*)(ws + 18874368);   // 2048x1024 f16      (4 MB)
  _Float16* qkvh  = (_Float16*)(ws + 23068672);   // 3 x [B][H][T][64]  (24 MB)
  _Float16* rh    = (_Float16*)(ws + 48234496);   // [H][T][64] f16     (4 MB)
  _Float16* attn  = (_Float16*)(ws + 52428800);   // 4096x1024 f16      (8 MB)

  cvt_kernel<<<(DMODEL * 3 * DMODEL + 255) / 256, 256, 0, stream>>>(Wqkv, wqkvh, DMODEL * 3 * DMODEL);
  cvt_kernel<<<(DMODEL * DMODEL + 255) / 256, 256, 0, stream>>>(Wr, wrh, DMODEL * DMODEL);
  cvt_kernel<<<(DMODEL * DMODEL + 255) / 256, 256, 0, stream>>>(Wo, woh, DMODEL * DMODEL);
  cvt_kernel<<<(T_SEQ * DMODEL + 255) / 256, 256, 0, stream>>>(pos, posh, T_SEQ * DMODEL);

  ln_kernel<<<T_SEQ * BATCH, 256, 0, stream>>>(x, ln_g, ln_b, xln);

  gemm_qkv_kernel<<<dim3((T_SEQ * BATCH) / 64, (3 * DMODEL) / 64), 128, 0, stream>>>(xln, wqkvh, qkvh);
  gemm_r_kernel<<<dim3(T_SEQ / 64, DMODEL / 64), 128, 0, stream>>>(posh, wrh, rh);

  const _Float16* qh = qkvh;
  const _Float16* kh = qkvh + (size_t)BATCH * NHEAD * T_SEQ * DHEAD;
  const _Float16* vh = kh   + (size_t)BATCH * NHEAD * T_SEQ * DHEAD;
  attn_kernel<<<dim3(T_SEQ / 16, NHEAD, BATCH), 32, 0, stream>>>(qh, kh, vh, rh, rwb, rrb, attn);

  gemm_out_kernel<<<dim3((T_SEQ * BATCH) / 64, DMODEL / 64), 128, 0, stream>>>(attn, woh, (float*)d_out);
}
